// MultiHeadAttention_31129922961455
// MI455X (gfx1250) — matrix operand order repaired and checked by reading, not compiled, not run on hardware
//
#include <hip/hip_runtime.h>

// ---------------------------------------------------------------------------
// MHA for MI455X (gfx1250, wave32, WMMA 16x16x32 f16 -> f32)
// B=2, S=2048, D=1024, H=16, DK=64
// ---------------------------------------------------------------------------

#define B_ 2
#define S_ 2048
#define D_ 1024
#define H_ 16
#define DK_ 64
#define NTOK (B_ * S_)                    // 4096
#define NE ((size_t)B_ * H_ * S_ * DK_)   // 4,194,304 elements per f16 buffer

typedef __attribute__((ext_vector_type(16))) _Float16 v16h;
typedef __attribute__((ext_vector_type(8)))  float    v8f;

union Frag { v16h v; unsigned u[8]; };

__device__ __forceinline__ unsigned pack2(float a, float b) {
    union { _Float16 h[2]; unsigned u; } x;
    x.h[0] = (_Float16)a; x.h[1] = (_Float16)b;
    return x.u;
}

// A-fragment K index (16-bit A, 16x32): lane-half picks +16
__device__ __forceinline__ int kA(int hi, int j) {
    return hi * 16 + 2 * j;
}
// B-fragment K index (16-bit B, 32x16): lane-half picks +16
__device__ __forceinline__ int kB(int hi, int j) {
    return hi * 16 + 2 * j;
}

// low 32 bits of a generic LDS pointer == wave-relative LDS byte offset
__device__ __forceinline__ unsigned lds_off(const void* p) {
    return (unsigned)(unsigned long long)p;
}

// ---------------------------------------------------------------------------
// Kernel 1: projection GEMM  y = X(4096x1024) * W^T + b  -> f16 head layout
//   MODE 0: Q, scaled by 1/8, layout [bh][s][dk]
//   MODE 1: K,                layout [bh][s][dk]
//   MODE 2: V, transposed     layout [bh][dk][s]
// block = 128 threads (4 waves); block tile 64(M) x 64(N); K chunk 32
// ---------------------------------------------------------------------------
template <int MODE>
__global__ __launch_bounds__(128) void proj_kernel(
    const float* __restrict__ X, const float* __restrict__ W,
    const float* __restrict__ bias, _Float16* __restrict__ out)
{
    const int tid  = threadIdx.x;
    const int lane = tid & 31;
    const int wv   = tid >> 5;
    const int hi   = (lane >> 4) & 1;
    const int l16  = lane & 15;
    const int m0   = blockIdx.x * 64;
    const int n0   = blockIdx.y * 64;

    __shared__ _Float16 Al[64 * 32];
    __shared__ _Float16 Bl[64 * 32];

    v8f acc[4] = {v8f(0.f), v8f(0.f), v8f(0.f), v8f(0.f)};

    for (int k0 = 0; k0 < D_; k0 += 32) {
        // stage A (activations, fp32 -> f16): 64 rows x 32
#pragma unroll
        for (int i = 0; i < 4; ++i) {
            int ft = tid + i * 128;             // 512 float4 total
            int row = ft >> 3, c4 = (ft & 7) * 4;
            const float4 xv = *(const float4*)(X + (size_t)(m0 + row) * D_ + k0 + c4);
            unsigned* dst = (unsigned*)(Al + row * 32 + c4);
            dst[0] = pack2(xv.x, xv.y);
            dst[1] = pack2(xv.z, xv.w);
        }
        // stage W rows n0..n0+63 (fp32 -> f16): 64 x 32, [n][k]
#pragma unroll
        for (int i = 0; i < 4; ++i) {
            int ft = tid + i * 128;
            int row = ft >> 3, c4 = (ft & 7) * 4;
            const float4 wvv = *(const float4*)(W + (size_t)(n0 + row) * D_ + k0 + c4);
            unsigned* dst = (unsigned*)(Bl + row * 32 + c4);
            dst[0] = pack2(wvv.x, wvv.y);
            dst[1] = pack2(wvv.z, wvv.w);
        }
        __syncthreads();

        // preload all fragments, then issue WMMAs back-to-back
        Frag a, bfr[4];
        {
            const unsigned* p = (const unsigned*)(Al + (16 * wv + l16) * 32);
#pragma unroll
            for (int j = 0; j < 8; ++j) a.u[j] = p[kA(hi, j) >> 1];
        }
#pragma unroll
        for (int nt = 0; nt < 4; ++nt) {
            const unsigned* p = (const unsigned*)(Bl + (16 * nt + l16) * 32);
#pragma unroll
            for (int j = 0; j < 8; ++j) bfr[nt].u[j] = p[kB(hi, j) >> 1];
        }
#pragma unroll
        for (int nt = 0; nt < 4; ++nt)
            acc[nt] = __builtin_amdgcn_wmma_f32_16x16x32_f16(
                false, a.v, false, bfr[nt].v, (short)0, acc[nt], false, false);
        __syncthreads();
    }

    // epilogue: bias + scatter into head layout (compile-time mode)
#pragma unroll
    for (int nt = 0; nt < 4; ++nt) {
        const int n  = n0 + 16 * nt + l16;
        const int h  = n >> 6, dk = n & 63;
        const float bv = bias[n];
        if (MODE == 2) {
            // V transposed: consecutive r -> consecutive s -> pack dword stores
            const int mbase = m0 + 16 * wv + 8 * hi;
            const int b = mbase >> 11, s = mbase & (S_ - 1);
            _Float16* orow = out + ((size_t)(b * H_ + h) * DK_ + dk) * S_ + s;
#pragma unroll
            for (int j2 = 0; j2 < 4; ++j2)
                *(unsigned*)(orow + 2 * j2) =
                    pack2(acc[nt][2 * j2] + bv, acc[nt][2 * j2 + 1] + bv);
        } else {
#pragma unroll
            for (int r = 0; r < 8; ++r) {
                const int m = m0 + 16 * wv + r + 8 * hi;      // token row
                const int b = m >> 11, s = m & (S_ - 1);
                float val = acc[nt][r] + bv;
                if (MODE == 0) val *= 0.125f;                 // 1/sqrt(DK)
                out[((size_t)(b * H_ + h) * S_ + s) * DK_ + dk] = (_Float16)val;
            }
        }
    }
}

// ---------------------------------------------------------------------------
// Kernel 2: causal flash attention, transposed formulation, async double-buffer.
// grid = (B*H, S/64); block = 128 (4 waves, 16 query rows each)
// K/V chunks staged LDS via GLOBAL_LOAD_ASYNC_TO_LDS_B128 (ASYNCcnt pipelined).
// Computes S^T = K * Q^T  (lane = query), online softmax per lane,
// O^T += V^T * P^T, then writes O row-major f16 [token][D].
// ---------------------------------------------------------------------------
__global__ __launch_bounds__(128) void attn_kernel(
    const _Float16* __restrict__ Qh, const _Float16* __restrict__ Kh,
    const _Float16* __restrict__ Vt, _Float16* __restrict__ O)
{
    const int bh = blockIdx.x;
    const int q0 = blockIdx.y * 64;
    const int b  = bh >> 4, h = bh & 15;
    const int tid  = threadIdx.x;
    const int lane = tid & 31;
    const int wv   = tid >> 5;
    const int hi   = (lane >> 4) & 1;
    const int l16  = lane & 15;

    __shared__ _Float16 Kl[2][64 * 64];   // [buf][key][feature]
    __shared__ _Float16 Vl[2][64 * 64];   // [buf][dk][key]

    const size_t baseQK = (size_t)bh * S_ * DK_;
    const int qg = q0 + 16 * wv + l16;           // this lane's query row

    const unsigned klBase = lds_off(&Kl[0][0]);
    const unsigned vlBase = lds_off(&Vl[0][0]);
    const int rowi = lane >> 3;                  // 0..3
    const int colb = (lane & 7) * 16;            // byte column within 128B row

    // issue 8 async b128 loads (per wave) for one 64-key chunk into buffer `buf`
    auto issue_chunk = [&](int jc, int buf) {
#pragma unroll
        for (int i = 0; i < 4; ++i) {
            const int row = 16 * wv + 4 * i + rowi;            // key row / dk row
            const unsigned ldsK = klBase + (unsigned)(buf * 8192 + row * 128 + colb);
            const char* gK = (const char*)(Kh + baseQK + (size_t)(jc * 64 + row) * DK_) + colb;
            asm volatile("global_load_async_to_lds_b128 %0, %1, off"
                         :: "v"(ldsK), "v"(gK) : "memory");
            const unsigned ldsV = vlBase + (unsigned)(buf * 8192 + row * 128 + colb);
            const char* gV = (const char*)(Vt + (size_t)(bh * DK_ + row) * S_ + jc * 64) + colb;
            asm volatile("global_load_async_to_lds_b128 %0, %1, off"
                         :: "v"(ldsV), "v"(gV) : "memory");
        }
    };

    // Q^T B-fragments (lane = query, regs = feature), loaded once from global
    Frag qf[2];
    {
        const _Float16* qrow = Qh + baseQK + (size_t)qg * DK_;
#pragma unroll
        for (int fc = 0; fc < 2; ++fc)
#pragma unroll
            for (int j = 0; j < 8; ++j)
                qf[fc].u[j] = *(const unsigned*)(qrow + fc * 32 + kB(hi, j));
    }

    v8f oacc[4] = {v8f(0.f), v8f(0.f), v8f(0.f), v8f(0.f)};
    float m_run = -3.0e38f, l_run = 0.0f;
    const int jlast = q0 >> 6;

    issue_chunk(0, 0);

    for (int jc = 0; jc <= jlast; ++jc) {
        const int k0  = jc * 64;
        const int cur = jc & 1;

        if (jc < jlast) {
            issue_chunk(jc + 1, 1 - cur);              // prefetch next chunk
            asm volatile("s_wait_asynccnt 0x8" ::: "memory");  // chunk jc resident
        } else {
            asm volatile("s_wait_asynccnt 0x0" ::: "memory");
        }
        __syncthreads();                               // all waves' loads visible

        // S^T tiles: 4 key-tiles x (2 feature chunks); preload 2 tiles ahead
        v8f sacc[4] = {v8f(0.f), v8f(0.f), v8f(0.f), v8f(0.f)};
#pragma unroll
        for (int tp = 0; tp < 2; ++tp) {
            Frag ka0, ka1, kb0, kb1;
            const unsigned* pa = (const unsigned*)(Kl[cur] + (32 * tp + l16) * 64);
            const unsigned* pb = (const unsigned*)(Kl[cur] + (32 * tp + 16 + l16) * 64);
#pragma unroll
            for (int j = 0; j < 8; ++j) {
                const int o = kA(hi, j) >> 1;
                ka0.u[j] = pa[o];  ka1.u[j] = pa[o + 16];
                kb0.u[j] = pb[o];  kb1.u[j] = pb[o + 16];
            }
            sacc[2 * tp] = __builtin_amdgcn_wmma_f32_16x16x32_f16(
                false, ka0.v, false, qf[0].v, (short)0, sacc[2 * tp], false, false);
            sacc[2 * tp] = __builtin_amdgcn_wmma_f32_16x16x32_f16(
                false, ka1.v, false, qf[1].v, (short)0, sacc[2 * tp], false, false);
            sacc[2 * tp + 1] = __builtin_amdgcn_wmma_f32_16x16x32_f16(
                false, kb0.v, false, qf[0].v, (short)0, sacc[2 * tp + 1], false, false);
            sacc[2 * tp + 1] = __builtin_amdgcn_wmma_f32_16x16x32_f16(
                false, kb1.v, false, qf[1].v, (short)0, sacc[2 * tp + 1], false, false);
        }

        // causal mask: only the diagonal chunk needs it
        if (jc == jlast) {
#pragma unroll
            for (int t = 0; t < 4; ++t)
#pragma unroll
                for (int r = 0; r < 8; ++r) {
                    int key = k0 + 16 * t + 8 * hi + r;
                    if (key > qg) sacc[t][r] = -1.0e30f;
                }
        }

        // online softmax (per-lane scalars; one shuffle merges half-waves)
        float mloc = -3.0e38f;
#pragma unroll
        for (int t = 0; t < 4; ++t)
#pragma unroll
            for (int r = 0; r < 8; ++r) mloc = fmaxf(mloc, sacc[t][r]);
        mloc = fmaxf(mloc, __shfl_xor(mloc, 16));
        const float mnew = fmaxf(m_run, mloc);
        const float corr = __expf(m_run - mnew);

        float lloc = 0.0f;
        unsigned up[4][4];
#pragma unroll
        for (int t = 0; t < 4; ++t)
#pragma unroll
            for (int j2 = 0; j2 < 4; ++j2) {
                float p0 = __expf(sacc[t][2 * j2]     - mnew);
                float p1 = __expf(sacc[t][2 * j2 + 1] - mnew);
                lloc += p0 + p1;
                up[t][j2] = pack2(p0, p1);
            }
        lloc += __shfl_xor(lloc, 16);
        l_run = l_run * corr + lloc;
        m_run = mnew;
#pragma unroll
        for (int td = 0; td < 4; ++td)
#pragma unroll
            for (int r = 0; r < 8; ++r) oacc[td][r] *= corr;

        // O^T += V^T * P^T per 32-key chunk. Issue the V^T fragment DS loads
        // first so their latency overlaps the P^T bpermute/select construction.
#pragma unroll
        for (int c = 0; c < 2; ++c) {
            Frag vf[4];
#pragma unroll
            for (int td = 0; td < 4; ++td) {
                const unsigned* p = (const unsigned*)(Vl[cur] + (16 * td + l16) * 64 + 32 * c);
#pragma unroll
                for (int j = 0; j < 8; ++j) vf[td].u[j] = p[kA(hi, j) >> 1];
            }
            Frag pf;
#pragma unroll
            for (int j2 = 0; j2 < 4; ++j2) {
                unsigned send = hi ? up[2 * c][j2] : up[2 * c + 1][j2];
                unsigned recv = (unsigned)__shfl_xor((int)send, 16);
                pf.u[j2]     = hi ? recv              : up[2 * c][j2];
                pf.u[4 + j2] = hi ? up[2 * c + 1][j2] : recv;
            }
#pragma unroll
            for (int td = 0; td < 4; ++td)
                oacc[td] = __builtin_amdgcn_wmma_f32_16x16x32_f16(
                    false, vf[td].v, false, pf.v, (short)0, oacc[td], false, false);
        }
        __syncthreads();   // buffer `cur` free for reuse at jc+2
    }

    // epilogue: normalize and store O row-major f16 [token][D]
    const float inv_l = 1.0f / l_run;
    const size_t obase = ((size_t)(b * S_ + qg)) * D_ + h * DK_;
#pragma unroll
    for (int td = 0; td < 4; ++td)
#pragma unroll
        for (int j2 = 0; j2 < 4; ++j2)
            *(unsigned*)(O + obase + 16 * td + 8 * hi + 2 * j2) =
                pack2(oacc[td][2 * j2] * inv_l, oacc[td][2 * j2 + 1] * inv_l);
}

// ---------------------------------------------------------------------------
// Kernel 3: output projection  out = O(4096x1024,f16) * Wo^T + bo  (fp32 out)
// ---------------------------------------------------------------------------
__global__ __launch_bounds__(128) void outproj_kernel(
    const _Float16* __restrict__ Oin, const float* __restrict__ W,
    const float* __restrict__ bias, float* __restrict__ out)
{
    const int tid  = threadIdx.x;
    const int lane = tid & 31;
    const int wv   = tid >> 5;
    const int hi   = (lane >> 4) & 1;
    const int l16  = lane & 15;
    const int m0   = blockIdx.x * 64;
    const int n0   = blockIdx.y * 64;

    __shared__ _Float16 Al[64 * 32];
    __shared__ _Float16 Bl[64 * 32];

    v8f acc[4] = {v8f(0.f), v8f(0.f), v8f(0.f), v8f(0.f)};

    for (int k0 = 0; k0 < D_; k0 += 32) {
        // stage A (already f16): 64 rows x 32 = 256 uint4
#pragma unroll
        for (int i = 0; i < 2; ++i) {
            int idx = tid + i * 128;
            int row = idx >> 2, c = idx & 3;
            ((uint4*)Al)[row * 4 + c] =
                *(const uint4*)(Oin + (size_t)(m0 + row) * D_ + k0 + c * 8);
        }
        // stage W (fp32 -> f16)
#pragma unroll
        for (int i = 0; i < 4; ++i) {
            int ft = tid + i * 128;
            int row = ft >> 3, c4 = (ft & 7) * 4;
            const float4 wvv = *(const float4*)(W + (size_t)(n0 + row) * D_ + k0 + c4);
            unsigned* dst = (unsigned*)(Bl + row * 32 + c4);
            dst[0] = pack2(wvv.x, wvv.y);
            dst[1] = pack2(wvv.z, wvv.w);
        }
        __syncthreads();

        Frag a, bfr[4];
        {
            const unsigned* p = (const unsigned*)(Al + (16 * wv + l16) * 32);
#pragma unroll
            for (int j = 0; j < 8; ++j) a.u[j] = p[kA(hi, j) >> 1];
        }
#pragma unroll
        for (int nt = 0; nt < 4; ++nt) {
            const unsigned* p = (const unsigned*)(Bl + (16 * nt + l16) * 32);
#pragma unroll
            for (int j = 0; j < 8; ++j) bfr[nt].u[j] = p[kB(hi, j) >> 1];
        }
#pragma unroll
        for (int nt = 0; nt < 4; ++nt)
            acc[nt] = __builtin_amdgcn_wmma_f32_16x16x32_f16(
                false, a.v, false, bfr[nt].v, (short)0, acc[nt], false, false);
        __syncthreads();
    }

#pragma unroll
    for (int nt = 0; nt < 4; ++nt) {
        const int n = n0 + 16 * nt + l16;
        const float bv = bias[n];
#pragma unroll
        for (int r = 0; r < 8; ++r) {
            const int m = m0 + 16 * wv + r + 8 * hi;
            out[(size_t)m * D_ + n] = acc[nt][r] + bv;
        }
    }
}

// ---------------------------------------------------------------------------
extern "C" void kernel_launch(void* const* d_in, const int* in_sizes, int n_in,
                              void* d_out, int out_size, void* d_ws, size_t ws_size,
                              hipStream_t stream) {
    const float* q = (const float*)d_in[0];
    const float* k = (const float*)d_in[1];
    const float* v = (const float*)d_in[2];
    // d_in[3] = mask (causal tril) -- enforced analytically in attn_kernel
    const float* Wq = (const float*)d_in[4];
    const float* bq = (const float*)d_in[5];
    const float* Wk = (const float*)d_in[6];
    const float* bk = (const float*)d_in[7];
    const float* Wv = (const float*)d_in[8];
    const float* bv = (const float*)d_in[9];
    const float* Wo = (const float*)d_in[10];
    const float* bo = (const float*)d_in[11];

    _Float16* ws = (_Float16*)d_ws;
    _Float16* Qh = ws;
    _Float16* Kh = ws + NE;
    _Float16* Vt = ws + 2 * NE;
    _Float16* O  = ws + 3 * NE;

    dim3 gGemm(NTOK / 64, D_ / 64);   // (64, 16)
    dim3 blk(128);

    proj_kernel<0><<<gGemm, blk, 0, stream>>>(q, Wq, bq, Qh);
    proj_kernel<1><<<gGemm, blk, 0, stream>>>(k, Wk, bk, Kh);
    proj_kernel<2><<<gGemm, blk, 0, stream>>>(v, Wv, bv, Vt);

    dim3 gAttn(B_ * H_, S_ / 64);     // (32, 32)
    attn_kernel<<<gAttn, blk, 0, stream>>>(Qh, Kh, Vt, O);

    outproj_kernel<<<gGemm, blk, 0, stream>>>(O, Wo, bo, (float*)d_out);
}
